// TurboQuantMSE_63797444215185
// MI455X (gfx1250) — matrix operand order repaired and checked by reading, not compiled, not run on hardware
//
#include <hip/hip_runtime.h>

typedef __attribute__((ext_vector_type(16))) _Float16 v16h;
typedef __attribute__((ext_vector_type(8)))  _Float16 v8h;
typedef __attribute__((ext_vector_type(8)))  float    v8f;

#define N 128
#define ROWS_PER_WG 32
#define THREADS 256
#define EPSF 1e-8f
#define INV_SQRT_N 0.08838834764831845f /* 1/sqrt(128) */
#define CELLS 4096

// ws layout: [0,4096) u8 lut | 4096 float lo | 4100 float invw | 4104 float maxc
//            | [4112, 4112+255*4) float bnd
#define WS_PAR_OFF 4096
#define WS_BND_OFF 4112

// Full binary search (build kernel only): searchsorted(bnd, v, 'left'), 255 bnds.
__device__ __forceinline__ int ssearch(const float* __restrict__ bnd, float v) {
    int idx = 0;
#pragma unroll
    for (int step = 128; step > 0; step >>= 1) {
        idx += (bnd[idx + step - 1] < v) ? step : 0;
    }
    return idx;
}

// LUT-accelerated searchsorted: lut[cell] = count(bnd < cellLeft); at most one
// boundary per cell (cell width << min Lloyd-Max spacing), 2 corrections for
// safety. bnd[255] is +INF sentinel so no bounds checks are needed.
__device__ __forceinline__ int qsearch(const unsigned char* __restrict__ lut,
                                       const float* __restrict__ bnd,
                                       float lo, float invw, float v) {
    float cf = fminf(fmaxf((v - lo) * invw, 0.0f), (float)(CELLS - 1));
    int idx = lut[(int)cf];
    idx += (bnd[idx] < v);
    idx += (bnd[idx] < v);
    return idx;
}

__global__ __launch_bounds__(256) void
build_lut_kernel(const float* __restrict__ centroids, unsigned char* __restrict__ ws)
{
    __shared__ float s_c[256];
    __shared__ float s_bnd[255];
    __shared__ float s_lo, s_w;
    const int t = threadIdx.x;
    float* par = (float*)(ws + WS_PAR_OFF);
    float* bnd = (float*)(ws + WS_BND_OFF);

    s_c[t] = centroids[t];
    __syncthreads();
    if (t < 255) s_bnd[t] = 0.5f * (s_c[t] + s_c[t + 1]);
    __syncthreads();
    if (t == 0) {
        float lo = s_bnd[0], hi = s_bnd[254];
        float w  = (hi - lo) / (float)CELLS;
        s_lo = lo; s_w = w;
        par[0] = lo;
        par[1] = 1.0f / w;
        float m = s_c[0];
        for (int i = 1; i < 256; ++i) m = fmaxf(m, s_c[i]);
        par[2] = m;
    }
    if (t < 255) bnd[t] = s_bnd[t];
    __syncthreads();
    const float lo = s_lo, w = s_w;
    for (int c = t; c < CELLS; c += 256) {
        float left = lo + w * (float)c;
        ws[c] = (unsigned char)ssearch(s_bnd, left);
    }
}

__global__ __launch_bounds__(THREADS) void
turboquant_kernel(const float* __restrict__ x,
                  const float* __restrict__ signs,
                  const float* __restrict__ centroids,
                  const unsigned char* __restrict__ ws,
                  float* __restrict__ out,
                  int nrows)
{
    __shared__ __align__(16) _Float16 s_xh[ROWS_PER_WG][N]; // f16 A staging (x*sign/norm, later recon)
    __shared__ __align__(16) float    s_rot[ROWS_PER_WG][N];// rotated tile (f32), reused for output
    __shared__ __align__(16) unsigned char s_lut[CELLS];
    __shared__ float s_c[256];
    __shared__ float s_bnd[256];                            // [255] = +INF sentinel
    __shared__ float s_signs[N];
    __shared__ float s_pA[ROWS_PER_WG][8];
    __shared__ float s_pB[ROWS_PER_WG][8];
    __shared__ float s_norm[ROWS_PER_WG];
    __shared__ float s_invnorm[ROWS_PER_WG];
    __shared__ float s_maxabs[ROWS_PER_WG];
    __shared__ float s_meanabs[ROWS_PER_WG];
    __shared__ float s_g1[ROWS_PER_WG];
    __shared__ float s_gamma[ROWS_PER_WG];
    __shared__ int   s_spiky[ROWS_PER_WG];

    const int t    = threadIdx.x;
    const int lane = t & 31;
    const int wave = t >> 5;
    const int mb   = wave >> 2;          // M block 0/1 (rows mb*16..mb*16+15)
    const int r    = t >> 3;             // row 0..31 (8 threads per row)
    const int cb   = t & 7;              // column chunk 0..7
    const int c0   = cb * 16;
    const long rowBase = (long)blockIdx.x * ROWS_PER_WG;
    if (rowBase >= nrows) return;

    const float* par = (const float*)(ws + WS_PAR_OFF);
    const float lo   = par[0];
    const float invw = par[1];
    const float maxc = par[2];

    // ---- stage tables + x (kept in registers), per-thread partial sumsq ----
    s_c[t] = centroids[t];
    if (t < N) s_signs[t] = signs[t];
    if (t < 255) s_bnd[t] = ((const float*)(ws + WS_BND_OFF))[t];
    else if (t == 255) s_bnd[255] = 3.0e38f;
    ((uint4*)s_lut)[t] = ((const uint4*)ws)[t];

    float xa[16];
    {
        const float4* xr = (const float4*)(x + (rowBase + r) * N + c0);
        float ss = 0.f;
#pragma unroll
        for (int q = 0; q < 4; ++q) {
            float4 vv = xr[q];
            xa[4*q+0] = vv.x; xa[4*q+1] = vv.y; xa[4*q+2] = vv.z; xa[4*q+3] = vv.w;
            ss += vv.x*vv.x + vv.y*vv.y + vv.z*vv.z + vv.w*vv.w;
        }
        s_pA[r][cb] = ss;
    }
    __syncthreads();

    if (t < ROWS_PER_WG) {
        float s = 0.f;
#pragma unroll
        for (int j = 0; j < 8; ++j) s += s_pA[t][j];
        float nm = sqrtf(s) + EPSF;
        s_norm[t] = nm;
        s_invnorm[t] = 1.0f / nm;
    }
    __syncthreads();

    // ---- stage A operand as f16: (x * signs) / norm ----
    {
        const float inv = s_invnorm[r];
        v8h h0, h1;
#pragma unroll
        for (int j = 0; j < 8; ++j) h0[j] = (_Float16)(xa[j]     * s_signs[c0 + j]     * inv);
#pragma unroll
        for (int j = 0; j < 8; ++j) h1[j] = (_Float16)(xa[8 + j] * s_signs[c0 + 8 + j] * inv);
        *(v8h*)&s_xh[r][c0]     = h0;
        *(v8h*)&s_xh[r][c0 + 8] = h1;
    }

    // ---- B fragments (Hadamard +-1), built ONCE via sign-bit packing ----
    v16h bfrag[2][4];
    {
        const unsigned kl = lane & 15;
        unsigned P0 = 0;
        if (kl & 1) P0 ^= 0xAAAAu;
        if (kl & 2) P0 ^= 0xCCCCu;
        if (kl & 4) P0 ^= 0xF0F0u;
        if (kl & 8) P0 ^= 0xFF00u;

        unsigned W[8];
        {
            unsigned Q = P0;
#pragma unroll
            for (int v = 0; v < 8; ++v) {
                W[v] = 0x3C003C00u | ((Q & 1u) << 15) | ((Q & 2u) << 30);
                Q >>= 2;
            }
        }
#pragma unroll
        for (int nt = 0; nt < 2; ++nt) {
            const int nbase = ((wave & 3) * 2 + nt) * 16;
#pragma unroll
            for (int kc = 0; kc < 4; ++kc) {
                const int kg = kc * 32 + lane;
                const unsigned fm = (__popc(kg & nbase) & 1) ? 0x80008000u : 0u;
                union { unsigned u[8]; v16h h; } bu;
#pragma unroll
                for (int v = 0; v < 8; ++v) bu.u[v] = W[v] ^ fm;
                bfrag[nt][kc] = bu.h;
            }
        }
    }
    __syncthreads();

    // ---- GEMM 1: rot = A @ H128 * (1/sqrt(N)) via WMMA ----
    {
        const int m   = mb * 16 + (lane & 15);
        const int khi = (lane >> 4) * 16;
        v16h afrag[4];
#pragma unroll
        for (int kc = 0; kc < 4; ++kc) {
            v8h lo8 = *(const v8h*)&s_xh[m][kc * 32 + khi];
            v8h hi8 = *(const v8h*)&s_xh[m][kc * 32 + khi + 8];
            afrag[kc] = __builtin_shufflevector(lo8, hi8, 0,1,2,3,4,5,6,7,8,9,10,11,12,13,14,15);
        }
#pragma unroll
        for (int nt = 0; nt < 2; ++nt) {
            const int nbase = ((wave & 3) * 2 + nt) * 16;
            v8f acc = {};
#pragma unroll
            for (int kc = 0; kc < 4; ++kc)
                acc = __builtin_amdgcn_wmma_f32_16x16x32_f16(
                    false, afrag[kc], false, bfrag[nt][kc], (short)0, acc, false, false);
            const int mrow = mb * 16 + ((lane < 16) ? 0 : 8);
            const int col  = nbase + (lane & 15);
#pragma unroll
            for (int i = 0; i < 8; ++i)
                s_rot[mrow + i][col] = acc[i] * INV_SQRT_N;
        }
    }
    __syncthreads();

    // ---- read back this thread's 16 rot values once; per-row max/mean ----
    float xv[16];
    {
        float mx = 0.f, sm = 0.f;
#pragma unroll
        for (int q = 0; q < 4; ++q) {
            float4 vv = ((const float4*)&s_rot[r][c0])[q];
            xv[4*q+0] = vv.x; xv[4*q+1] = vv.y; xv[4*q+2] = vv.z; xv[4*q+3] = vv.w;
        }
#pragma unroll
        for (int j = 0; j < 16; ++j) { float a = fabsf(xv[j]); mx = fmaxf(mx, a); sm += a; }
        s_pA[r][cb] = mx;
        s_pB[r][cb] = sm;
    }
    __syncthreads();
    if (t < ROWS_PER_WG) {
        float M = 0.f, S = 0.f;
#pragma unroll
        for (int j = 0; j < 8; ++j) { M = fmaxf(M, s_pA[t][j]); S += s_pB[t][j]; }
        s_maxabs[t]  = M;
        s_meanabs[t] = S * (1.0f / (float)N) + EPSF;
    }
    __syncthreads();

    // ---- pass 1: quantize with rms scale, cache idx1, fit g1 ----
    int i1[16], i2[16];
    {
        const float rms = s_maxabs[r] / maxc;
        const float inv = 1.0f / (rms + EPSF);
        float da = 0.f, db = 0.f;
#pragma unroll
        for (int j = 0; j < 16; ++j) {
            int id = qsearch(s_lut, s_bnd, lo, invw, xv[j] * inv);
            i1[j] = id;
            float rv = s_c[id];
            da += xv[j] * rv;
            db += rv * rv;
        }
        s_pA[r][cb] = da;
        s_pB[r][cb] = db;
    }
    __syncthreads();
    if (t < ROWS_PER_WG) {
        float DA = 0.f, DB = 0.f;
#pragma unroll
        for (int j = 0; j < 8; ++j) { DA += s_pA[t][j]; DB += s_pB[t][j]; }
        s_g1[t] = DA / (DB + EPSF);
    }
    __syncthreads();

    // ---- pass 2: quantize with g1, cache idx2, fit g2; pick gamma/index ----
    {
        const float inv = 1.0f / (s_g1[r] + EPSF);
        float da = 0.f, db = 0.f;
#pragma unroll
        for (int j = 0; j < 16; ++j) {
            int id = qsearch(s_lut, s_bnd, lo, invw, xv[j] * inv);
            i2[j] = id;
            float rv = s_c[id];
            da += xv[j] * rv;
            db += rv * rv;
        }
        s_pA[r][cb] = da;
        s_pB[r][cb] = db;
    }
    __syncthreads();
    if (t < ROWS_PER_WG) {
        float DA = 0.f, DB = 0.f;
#pragma unroll
        for (int j = 0; j < 8; ++j) { DA += s_pA[t][j]; DB += s_pB[t][j]; }
        float g2  = DA / (DB + EPSF);
        float rms = s_maxabs[t] / maxc;
        int spiky = (s_maxabs[t] / s_meanabs[t]) > 5.0f;
        s_spiky[t] = spiky;
        s_gamma[t] = spiky ? rms : g2;
    }
    __syncthreads();

    // ---- reconstruct from cached indices into f16 staging ----
    {
        const float gm = s_gamma[r];
        const int   sp = s_spiky[r];
        v8h h0, h1;
#pragma unroll
        for (int j = 0; j < 8; ++j)  h0[j] = (_Float16)(s_c[sp ? i1[j]     : i2[j]]     * gm);
#pragma unroll
        for (int j = 0; j < 8; ++j)  h1[j] = (_Float16)(s_c[sp ? i1[8 + j] : i2[8 + j]] * gm);
        *(v8h*)&s_xh[r][c0]     = h0;
        *(v8h*)&s_xh[r][c0 + 8] = h1;
    }
    __syncthreads();

    // ---- GEMM 2: xhat = (recon @ H128) * (1/sqrt(N)) * signs[col] * norm[row] ----
    {
        const int m   = mb * 16 + (lane & 15);
        const int khi = (lane >> 4) * 16;
        v16h afrag[4];
#pragma unroll
        for (int kc = 0; kc < 4; ++kc) {
            v8h lo8 = *(const v8h*)&s_xh[m][kc * 32 + khi];
            v8h hi8 = *(const v8h*)&s_xh[m][kc * 32 + khi + 8];
            afrag[kc] = __builtin_shufflevector(lo8, hi8, 0,1,2,3,4,5,6,7,8,9,10,11,12,13,14,15);
        }
#pragma unroll
        for (int nt = 0; nt < 2; ++nt) {
            const int nbase = ((wave & 3) * 2 + nt) * 16;
            v8f acc = {};
#pragma unroll
            for (int kc = 0; kc < 4; ++kc)
                acc = __builtin_amdgcn_wmma_f32_16x16x32_f16(
                    false, afrag[kc], false, bfrag[nt][kc], (short)0, acc, false, false);
            const int mrow = mb * 16 + ((lane < 16) ? 0 : 8);
            const int col  = nbase + (lane & 15);
            const float sc = INV_SQRT_N * s_signs[col];
#pragma unroll
            for (int i = 0; i < 8; ++i)
                s_rot[mrow + i][col] = acc[i] * sc * s_norm[mrow + i];
        }
    }
    __syncthreads();

    // ---- coalesced store ----
    {
        float4* orow = (float4*)(out + (rowBase + r) * N + c0);
#pragma unroll
        for (int q = 0; q < 4; ++q)
            orow[q] = ((const float4*)&s_rot[r][c0])[q];
    }
}

extern "C" void kernel_launch(void* const* d_in, const int* in_sizes, int n_in,
                              void* d_out, int out_size, void* d_ws, size_t ws_size,
                              hipStream_t stream) {
    const float* x         = (const float*)d_in[0];   // (2,32,4096,128) f32
    const float* signs     = (const float*)d_in[1];   // (128,) f32
    const float* centroids = (const float*)d_in[2];   // (256,) f32
    float* out             = (float*)d_out;
    unsigned char* ws      = (unsigned char*)d_ws;

    build_lut_kernel<<<1, 256, 0, stream>>>(centroids, ws);

    const int nrows = in_sizes[0] / N;                // 262144
    const int grid  = (nrows + ROWS_PER_WG - 1) / ROWS_PER_WG;
    turboquant_kernel<<<grid, THREADS, 0, stream>>>(x, signs, centroids, ws, out, nrows);
}
